// StableGNN_45904610460214
// MI455X (gfx1250) — compile-verified, matrix operand order checked by reading, not run on hardware
//
#include <hip/hip_runtime.h>
#include <hip/hip_bf16.h>

// ---------------------------------------------------------------------------
// StableGNN (3-layer GCN, eval mode) for MI455X / gfx1250.
// GEMMs use v_wmma_f32_16x16x32_bf16 (f32 accumulate); aggregation uses
// hardware global_atomic_add_f32; edge streams get global_prefetch_b8.
// ---------------------------------------------------------------------------

#define NNODES 50000
#define CDIM   128
#define HDIM   128
#define NEDGES 1600000
#define EPS_F  1e-5f
#define BN_EPS 1e-5f

typedef __bf16  bf16_t;
typedef __attribute__((ext_vector_type(16))) __bf16 v16bf;
typedef __attribute__((ext_vector_type(8)))  __bf16 v8bf;
typedef __attribute__((ext_vector_type(8)))  float  v8f;

// ---------------------------------------------------------------------------
// Precompute kernels
// ---------------------------------------------------------------------------

// deg[i] = 1.0 (self-loop weight), sumabs = 0
__global__ void init_deg_kernel(float* __restrict__ deg, float* __restrict__ sumabs) {
    int i = blockIdx.x * blockDim.x + threadIdx.x;
    if (i < NNODES) deg[i] = 1.0f;
    if (i == 0) sumabs[0] = 0.0f;
}

// sumabs += sum(|clip(ew, -3, 3)|)   (block-reduced, one atomic per block)
__global__ void sumabs_kernel(const float* __restrict__ ew, float* __restrict__ sumabs) {
    __shared__ float red[256];
    float acc = 0.0f;
    for (long long e = blockIdx.x * (long long)blockDim.x + threadIdx.x;
         e < NEDGES; e += (long long)gridDim.x * blockDim.x) {
        float w = fminf(fmaxf(ew[e], -3.0f), 3.0f);
        acc += fabsf(w);
    }
    red[threadIdx.x] = acc;
    __syncthreads();
    for (int s = 128; s > 0; s >>= 1) {
        if (threadIdx.x < s) red[threadIdx.x] += red[threadIdx.x + s];
        __syncthreads();
    }
    if (threadIdx.x == 0)
        __hip_atomic_fetch_add(sumabs, red[0], __ATOMIC_RELAXED, __HIP_MEMORY_SCOPE_AGENT);
}

// deg[col[e]] += w_eff(e)
__global__ void degree_kernel(const long long* __restrict__ col,
                              const float* __restrict__ ew,
                              const float* __restrict__ sumabs,
                              float* __restrict__ deg) {
    long long e = blockIdx.x * (long long)blockDim.x + threadIdx.x;
    if (e >= NEDGES) return;
    float w = fminf(fmaxf(ew[e], -3.0f), 3.0f);
    if (sumabs[0] < EPS_F) w = 1.0f;
    int c = (int)col[e];
    __hip_atomic_fetch_add(deg + c, w, __ATOMIC_RELAXED, __HIP_MEMORY_SCOPE_AGENT);
}

// dinv[i] = deg>0 ? rsqrt(max(deg,1e-12)) : 0 ;  selfnorm[i] = dinv^2
__global__ void dinv_kernel(const float* __restrict__ deg,
                            float* __restrict__ dinv,
                            float* __restrict__ selfnorm) {
    int i = blockIdx.x * blockDim.x + threadIdx.x;
    if (i >= NNODES) return;
    float d  = deg[i];
    float di = (d > 0.0f) ? rsqrtf(fmaxf(d, 1e-12f)) : 0.0f;
    dinv[i]     = di;
    selfnorm[i] = di * di;
}

// norm[e] = dinv[row]*w_eff*dinv[col]
__global__ void norm_kernel(const long long* __restrict__ row,
                            const long long* __restrict__ col,
                            const float* __restrict__ ew,
                            const float* __restrict__ sumabs,
                            const float* __restrict__ dinv,
                            float* __restrict__ norm) {
    long long e = blockIdx.x * (long long)blockDim.x + threadIdx.x;
    if (e >= NEDGES) return;
    float w = fminf(fmaxf(ew[e], -3.0f), 3.0f);
    if (sumabs[0] < EPS_F) w = 1.0f;
    norm[e] = dinv[(int)row[e]] * w * dinv[(int)col[e]];
}

// act = bf16(clip(x, -10, 10))
__global__ void clipx_kernel(const float* __restrict__ x, bf16_t* __restrict__ act) {
    long long i = blockIdx.x * (long long)blockDim.x + threadIdx.x;
    if (i >= (long long)NNODES * CDIM) return;
    float v = fminf(fmaxf(x[i], -10.0f), 10.0f);
    act[i] = (bf16_t)v;
}

// Pack W [128(K) x 128(N)] fp32 row-major into bf16 WMMA B-fragments:
// dst[((ks*8 + nt)*32 + lane)*16 + j]  so a lane reads 32 contiguous bytes.
// Element j<8  -> K = ks*32 + kb + j        (kb = (lane/16)*8)
// Element j>=8 -> K = ks*32 + kb + 16 + (j-8)
// N = nt*16 + (lane & 15)
__global__ void pack_w_kernel(const float* __restrict__ W, bf16_t* __restrict__ dst) {
    int t = blockIdx.x * blockDim.x + threadIdx.x;
    if (t >= HDIM * HDIM) return;
    int j    = t & 15;
    int lane = (t >> 4) & 31;
    int nt   = (t >> 9) & 7;
    int ks   = (t >> 12) & 3;
    int n    = nt * 16 + (lane & 15);
    int kb   = (lane >> 4) * 8;
    int k    = ks * 32 + kb + ((j < 8) ? j : (j - 8 + 16));
    dst[t] = (bf16_t)W[k * HDIM + n];
}

// ---------------------------------------------------------------------------
// GEMM: Hout = A(bf16) @ Wfrag(bf16), f32 accumulate via WMMA.
// One wave per 16-row strip, all 8 column tiles. Epilogue also seeds the
// scatter buffer with the self-loop term: Agg = Hout * selfnorm[row].
// ---------------------------------------------------------------------------
__global__ __launch_bounds__(256)
void gemm_wmma_kernel(const bf16_t* __restrict__ A,      // [N,128] bf16
                      const bf16_t* __restrict__ Bfrag,  // packed fragments
                      float* __restrict__ Hout,          // [N,128]
                      float* __restrict__ Agg,           // [N,128] = Hout*selfnorm
                      const float* __restrict__ selfnorm) {
    const int lane = threadIdx.x & 31;
    const int rt   = blockIdx.x * 8 + (threadIdx.x >> 5);   // 16-row tile index
    if (rt >= NNODES / 16) return;

    const int mrow = rt * 16 + (lane & 15);      // A-fragment row for this lane
    const int kb   = (lane >> 4) * 8;            // 0 or 8

    v8f acc[8];
#pragma unroll
    for (int i = 0; i < 8; ++i) acc[i] = (v8f)(0.0f);

#pragma unroll
    for (int ks = 0; ks < 4; ++ks) {
        // A fragment: two aligned 16B loads, shuffled into the v16bf layout.
        const bf16_t* arow = A + (size_t)mrow * HDIM + ks * 32 + kb;
        v8bf lo = *(const v8bf*)(arow);
        v8bf hi = *(const v8bf*)(arow + 16);
        v16bf a = __builtin_shufflevector(lo, hi,
                  0, 1, 2, 3, 4, 5, 6, 7, 8, 9, 10, 11, 12, 13, 14, 15);
#pragma unroll
        for (int nt = 0; nt < 8; ++nt) {
            v16bf b = *(const v16bf*)(Bfrag + (((ks * 8 + nt) * 32) + lane) * 16);
            acc[nt] = __builtin_amdgcn_wmma_f32_16x16x32_bf16(
                false, a, false, b, (short)0, acc[nt], false, false);
        }
    }

    // Epilogue: C/D layout => reg v, lane l: row = rt*16 + v + (l/16)*8, col = nt*16 + l%16
    const int mbase = rt * 16 + (lane >> 4) * 8;
    const int ncol  = lane & 15;
#pragma unroll
    for (int v = 0; v < 8; ++v) {
        const int m = mbase + v;
        const float sn = selfnorm[m];
        float* hp = Hout + (size_t)m * HDIM + ncol;
        float* ap = Agg  + (size_t)m * HDIM + ncol;
#pragma unroll
        for (int nt = 0; nt < 8; ++nt) {
            float val = acc[nt][v];
            hp[nt * 16] = val;
            ap[nt * 16] = val * sn;
        }
    }
}

// ---------------------------------------------------------------------------
// Edge aggregation: one wave per edge, lane handles 4 features.
// Agg[col, :] += H[row, :] * norm[e]   (hardware f32 atomics, L2-resident)
// ---------------------------------------------------------------------------
__global__ void edge_scatter_kernel(const long long* __restrict__ row,
                                    const long long* __restrict__ col,
                                    const float* __restrict__ norm,
                                    const float* __restrict__ H,
                                    float* __restrict__ Agg) {
    long long tid = blockIdx.x * (long long)blockDim.x + threadIdx.x;
    long long e   = tid >> 5;
    if (e >= NEDGES) return;
    int lane = (int)(tid & 31);

    // Stream-ahead prefetch of edge metadata (lowers to global_prefetch_b8).
    if (lane == 0 && e + 4096 < NEDGES) {
        __builtin_prefetch(row + e + 4096, 0, 1);
        __builtin_prefetch(col + e + 4096, 0, 1);
        __builtin_prefetch(norm + e + 4096, 0, 1);
    }

    long long r = row[e];
    long long c = col[e];
    float     w = norm[e];

    const float4 hv = *(const float4*)(H + r * HDIM + lane * 4);
    float* dst = Agg + c * HDIM + lane * 4;
    __hip_atomic_fetch_add(dst + 0, hv.x * w, __ATOMIC_RELAXED, __HIP_MEMORY_SCOPE_AGENT);
    __hip_atomic_fetch_add(dst + 1, hv.y * w, __ATOMIC_RELAXED, __HIP_MEMORY_SCOPE_AGENT);
    __hip_atomic_fetch_add(dst + 2, hv.z * w, __ATOMIC_RELAXED, __HIP_MEMORY_SCOPE_AGENT);
    __hip_atomic_fetch_add(dst + 3, hv.w * w, __ATOMIC_RELAXED, __HIP_MEMORY_SCOPE_AGENT);
}

// ---------------------------------------------------------------------------
// Post: y = (Agg + bias) * (gamma/sqrt(1+eps)) + beta ; optional ReLU.
// Writes bf16 activations for the next GEMM and/or fp32 final output.
// ---------------------------------------------------------------------------
__global__ void post_bn_kernel(const float* __restrict__ Agg,
                               const float* __restrict__ bias,
                               const float* __restrict__ gamma,
                               const float* __restrict__ beta,
                               bf16_t* __restrict__ actOut,   // may be null
                               float* __restrict__ fOut,      // may be null
                               int doRelu) {
    long long i = blockIdx.x * (long long)blockDim.x + threadIdx.x;
    if (i >= (long long)NNODES * HDIM) return;
    int ch = (int)(i & (HDIM - 1));
    float s = gamma[ch] * rsqrtf(1.0f + BN_EPS);
    float v = (Agg[i] + bias[ch]) * s + beta[ch];
    if (doRelu) v = fmaxf(v, 0.0f);
    if (actOut) actOut[i] = (bf16_t)v;
    if (fOut)   fOut[i] = v;
}

// ---------------------------------------------------------------------------
// Host launcher
// ---------------------------------------------------------------------------
extern "C" void kernel_launch(void* const* d_in, const int* in_sizes, int n_in,
                              void* d_out, int out_size, void* d_ws, size_t ws_size,
                              hipStream_t stream) {
    (void)in_sizes; (void)n_in; (void)out_size; (void)ws_size;

    const float*     x   = (const float*)d_in[0];
    const long long* ei  = (const long long*)d_in[1];     // [2, E] int64
    const float*     ew  = (const float*)d_in[2];
    const float*     Wm[3]  = { (const float*)d_in[3], (const float*)d_in[5], (const float*)d_in[7] };
    const float*     bm[3]  = { (const float*)d_in[4], (const float*)d_in[6], (const float*)d_in[8] };
    const float*     gm[3]  = { (const float*)d_in[9],  (const float*)d_in[11], (const float*)d_in[13] };
    const float*     bem[3] = { (const float*)d_in[10], (const float*)d_in[12], (const float*)d_in[14] };
    float* out = (float*)d_out;

    const long long* row = ei;
    const long long* col = ei + NEDGES;

    // Workspace carve-out (256B aligned slices).
    size_t off = 0;
    auto carve = [&](size_t bytes) -> void* {
        void* p = (char*)d_ws + off;
        off += (bytes + 255) & ~(size_t)255;
        return p;
    };
    float*  norm     = (float*)carve((size_t)NEDGES * 4);
    float*  deg      = (float*)carve((size_t)NNODES * 4);
    float*  dinv     = (float*)carve((size_t)NNODES * 4);
    float*  selfnorm = (float*)carve((size_t)NNODES * 4);
    float*  hbuf     = (float*)carve((size_t)NNODES * HDIM * 4);
    float*  aggbuf   = (float*)carve((size_t)NNODES * HDIM * 4);
    bf16_t* act      = (bf16_t*)carve((size_t)NNODES * HDIM * 2);
    bf16_t* wf[3];
    for (int l = 0; l < 3; ++l) wf[l] = (bf16_t*)carve((size_t)HDIM * HDIM * 2);
    float*  sumabs   = (float*)carve(256);

    const int T = 256;
    const int nodeBlocks = (NNODES + T - 1) / T;
    const int edgeBlocks = (NEDGES + T - 1) / T;
    const long long elemTotal = (long long)NNODES * HDIM;
    const int elemBlocks = (int)((elemTotal + T - 1) / T);
    const int scatterBlocks = (int)(((long long)NEDGES * 32 + T - 1) / T);
    const int gemmBlocks = (NNODES / 16 + 7) / 8;   // 8 waves/block, 1 wave/tile

    // --- precompute ---
    init_deg_kernel<<<nodeBlocks, T, 0, stream>>>(deg, sumabs);
    sumabs_kernel<<<1024, T, 0, stream>>>(ew, sumabs);
    degree_kernel<<<edgeBlocks, T, 0, stream>>>(col, ew, sumabs, deg);
    dinv_kernel<<<nodeBlocks, T, 0, stream>>>(deg, dinv, selfnorm);
    norm_kernel<<<edgeBlocks, T, 0, stream>>>(row, col, ew, sumabs, dinv, norm);
    clipx_kernel<<<elemBlocks, T, 0, stream>>>(x, act);
    for (int l = 0; l < 3; ++l)
        pack_w_kernel<<<(HDIM * HDIM + T - 1) / T, T, 0, stream>>>(Wm[l], wf[l]);

    // --- 3 GCN layers ---
    for (int l = 0; l < 3; ++l) {
        gemm_wmma_kernel<<<gemmBlocks, T, 0, stream>>>(act, wf[l], hbuf, aggbuf, selfnorm);
        edge_scatter_kernel<<<scatterBlocks, T, 0, stream>>>(row, col, norm, hbuf, aggbuf);
        if (l < 2) {
            post_bn_kernel<<<elemBlocks, T, 0, stream>>>(aggbuf, bm[l], gm[l], bem[l],
                                                         act, (float*)nullptr, 1);
        } else {
            post_bn_kernel<<<elemBlocks, T, 0, stream>>>(aggbuf, bm[l], gm[l], bem[l],
                                                         (bf16_t*)nullptr, out, 0);
        }
    }
}
